// AlphatRNN_19241453486721
// MI455X (gfx1250) — matrix operand order repaired and checked by reading, not compiled, not run on hardware
//
#include <hip/hip_runtime.h>
#include <hip/hip_bf16.h>
#include <math.h>

typedef __attribute__((ext_vector_type(16))) __bf16 v16bf;
typedef __attribute__((ext_vector_type(8)))  float  v8f;

namespace {
constexpr int B = 128, T = 512, D = 256, H = 1024;
constexpr int K0 = D + H;      // 1280 : [x | s]  for layer 0
constexpr int K1 = 2 * H;      // 2048 : [h0 | s] for layer 1
constexpr size_t ST = (size_t)B * H;   // one state matrix

// ---- workspace layout (bytes). Total ~53.2 MB. ----
constexpr size_t WS_BAR = 0;                                              // barrier counter
constexpr size_t WS_W0A = 256;                                            // [H,K0] bf16 = [Wax0|Uas0]
constexpr size_t WS_W0H = WS_W0A + sizeof(unsigned short)*(size_t)H*K0;   // [H,K0] bf16 = [Wih0|Whh0]
constexpr size_t WS_W1A = WS_W0H + sizeof(unsigned short)*(size_t)H*K0;   // [H,K1] bf16 = [Wax1|Uas1]
constexpr size_t WS_W1H = WS_W1A + sizeof(unsigned short)*(size_t)H*K1;   // [H,K1] bf16 = [Wih1|Whh1]
constexpr size_t WS_XBF = WS_W1H + sizeof(unsigned short)*(size_t)H*K1;   // [B,T,D] bf16 input
constexpr size_t WS_STF = WS_XBF + sizeof(unsigned short)*(size_t)B*T*D;  // 8*ST f32: s0[2],h0[2],s1[2],h1[2]
constexpr size_t WS_STB = WS_STF + sizeof(float)*ST*8;                    // 8*ST bf16: same order
constexpr int NBLOCKS = 64, NTHREADS = 256;  // 512 wave32s == 512 output tiles
}

__device__ __forceinline__ unsigned short f2bf(float f) {
  unsigned u = __float_as_uint(f);
  u += 0x7fffu + ((u >> 16) & 1u);           // round-to-nearest-even
  return (unsigned short)(u >> 16);
}

union V16 { uint4 q[2]; v16bf v; };

__device__ __forceinline__ v8f wmma_bf16(const V16& a, const V16& b, v8f c) {
  return __builtin_amdgcn_wmma_f32_16x16x32_bf16(false, a.v, false, b.v,
                                                 (short)0, c, false, false);
}

// One contiguous K-span of a 16x16 tile GEMM on two accumulator chains.
// arow/wrow are this lane's base pointers (already include half-wave offsets).
// kn must be a multiple of 64 (true for all spans here: 256 or 1024).
__device__ __forceinline__ void gemm_span(v8f& acc0, v8f& acc1,
                                          const unsigned short* __restrict__ arow,
                                          const unsigned short* __restrict__ wrow,
                                          int kn)
{
  #pragma unroll 2
  for (int k = 0; k < kn; k += 64) {
    V16 a0, b0, a1, b1;
    a0.q[0] = *(const uint4*)(arow + k);          // elems 0..7  -> K = k+8*hi+{0..7}
    a0.q[1] = *(const uint4*)(arow + k + 16);     // elems 8..15 -> K = k+16+8*hi+{0..7}
    b0.q[0] = *(const uint4*)(wrow + k);          // K = k+8*hi+{0..7}
    b0.q[1] = *(const uint4*)(wrow + k + 16);     // K = k+16+8*hi+{0..7}
    a1.q[0] = *(const uint4*)(arow + k + 32);
    a1.q[1] = *(const uint4*)(arow + k + 48);
    b1.q[0] = *(const uint4*)(wrow + k + 32);
    b1.q[1] = *(const uint4*)(wrow + k + 48);
    acc0 = wmma_bf16(a0, b0, acc0);
    acc1 = wmma_bf16(a1, b1, acc1);
  }
}

// 16x16 tile of D = A(16xK) * W^T(Kx16), A = [A0 | A1] split at ksplit.
__device__ __forceinline__ v8f tile_gemm(const unsigned short* __restrict__ A0, int lda0,
                                         const unsigned short* __restrict__ A1, int lda1,
                                         int ksplit, int ktot,
                                         const unsigned short* __restrict__ W,
                                         int aRow0, int wRow0, int lane)
{
  const int m  = lane & 15;
  const int hi = (lane >> 4) & 1;
  const unsigned short* wrow  = W  + (size_t)(wRow0 + m) * ktot + 8 * hi;
  const unsigned short* a0row = A0 + (size_t)(aRow0 + m) * lda0 + 8 * hi;
  const unsigned short* a1row = A1 + (size_t)(aRow0 + m) * lda1 + 8 * hi;
  v8f acc0 = {0.f,0.f,0.f,0.f,0.f,0.f,0.f,0.f};
  v8f acc1 = {0.f,0.f,0.f,0.f,0.f,0.f,0.f,0.f};
  gemm_span(acc0, acc1, a0row, wrow, ksplit);                    // region 0
  gemm_span(acc0, acc1, a1row, wrow + ksplit, ktot - ksplit);    // region 1
  return acc0 + acc1;
}

// grid-wide barrier: monotonic counter, one arrival per block
__device__ __forceinline__ void grid_barrier(unsigned int* bar, unsigned int target) {
  __syncthreads();
  __threadfence();
  if (threadIdx.x == 0) {
    atomicAdd(bar, 1u);
    while (__hip_atomic_load(bar, __ATOMIC_RELAXED, __HIP_MEMORY_SCOPE_AGENT) < target)
      __builtin_amdgcn_s_sleep(1);
  }
  __syncthreads();
  __threadfence();
}

__global__ void __launch_bounds__(NTHREADS)
alphat_init(float* __restrict__ stf, unsigned short* __restrict__ stb,
            unsigned int* __restrict__ bar)
{
  size_t i = (size_t)blockIdx.x * blockDim.x + threadIdx.x;
  size_t stride = (size_t)gridDim.x * blockDim.x;
  for (size_t k = i; k < ST * 8; k += stride) { stf[k] = 0.f; stb[k] = 0; }
  if (i == 0) *bar = 0u;
}

__global__ void __launch_bounds__(NTHREADS)
alphat_convert_w(const float* __restrict__ Wax0, const float* __restrict__ Uas0,
                 const float* __restrict__ Wih0, const float* __restrict__ Whh0,
                 const float* __restrict__ Wax1, const float* __restrict__ Uas1,
                 const float* __restrict__ Wih1, const float* __restrict__ Whh1,
                 unsigned short* __restrict__ W0a, unsigned short* __restrict__ W0h,
                 unsigned short* __restrict__ W1a, unsigned short* __restrict__ W1h)
{
  size_t i = (size_t)blockIdx.x * blockDim.x + threadIdx.x;
  size_t stride = (size_t)gridDim.x * blockDim.x;
  for (size_t idx = i; idx < (size_t)H * K0; idx += stride) {
    int row = (int)(idx / K0), k = (int)(idx % K0);
    W0a[idx] = f2bf(k < D ? Wax0[(size_t)row * D + k] : Uas0[(size_t)row * H + (k - D)]);
    W0h[idx] = f2bf(k < D ? Wih0[(size_t)row * D + k] : Whh0[(size_t)row * H + (k - D)]);
  }
  for (size_t idx = i; idx < (size_t)H * K1; idx += stride) {
    int row = (int)(idx / K1), k = (int)(idx % K1);
    W1a[idx] = f2bf(k < H ? Wax1[(size_t)row * H + k] : Uas1[(size_t)row * H + (k - H)]);
    W1h[idx] = f2bf(k < H ? Wih1[(size_t)row * H + k] : Whh1[(size_t)row * H + (k - H)]);
  }
}

__global__ void __launch_bounds__(NTHREADS)
alphat_convert_x(const float* __restrict__ x, unsigned short* __restrict__ xbf, size_t n)
{
  size_t i = (size_t)blockIdx.x * blockDim.x + threadIdx.x;
  size_t stride = (size_t)gridDim.x * blockDim.x;
  for (size_t k = i; k < n; k += stride) xbf[k] = f2bf(x[k]);
}

__global__ void __launch_bounds__(NTHREADS)
alphat_rnn_persistent(const unsigned short* __restrict__ xbf,
                      const unsigned short* __restrict__ W0a,
                      const unsigned short* __restrict__ W0h,
                      const unsigned short* __restrict__ W1a,
                      const unsigned short* __restrict__ W1h,
                      const float* __restrict__ ba0, const float* __restrict__ bh0,
                      const float* __restrict__ ba1, const float* __restrict__ bh1,
                      float* __restrict__ stf, unsigned short* __restrict__ stb,
                      unsigned int* __restrict__ bar)
{
  const int lane  = threadIdx.x & 31;
  const int wave  = blockIdx.x * (NTHREADS >> 5) + (threadIdx.x >> 5); // 0..511
  const int tileM = wave & 7;     // batch tile  (B/16 = 8)
  const int tileN = wave >> 3;    // hidden tile (H/16 = 64); uniform within a block -> weight reuse in WGP$
  const int m  = lane & 15;
  const int hi = (lane >> 4) & 1;
  const int bRow0 = tileM * 16 + hi * 8;   // C/D layout: VGPR r -> row r (+8 for upper half-wave)
  const int hCol  = tileN * 16 + m;
  const int aR0 = tileM * 16, wR0 = tileN * 16;

  unsigned int target = 0;
  for (int t = 0; t < T; ++t) {
    const int cur = t & 1, nxt = cur ^ 1;
    const unsigned short* xrow = xbf + (size_t)t * D;          // row stride = T*D

    // ---- Phase 1: alpha0 = sigmoid([x_t|s0] W0a^T + ba0); s0' = a*h0 + (1-a)*s0 ----
    {
      v8f acc = tile_gemm(xrow, T * D, stb + (0 + cur) * ST, H, D, K0, W0a, aR0, wR0, lane);
      const float* h0c = stf + (2 + cur) * ST;
      const float* s0c = stf + (0 + cur) * ST;
      float*          s0n  = stf + (0 + nxt) * ST;
      unsigned short* s0nb = stb + (0 + nxt) * ST;
      const float bias = ba0[hCol];
      #pragma unroll
      for (int r = 0; r < 8; ++r) {
        size_t idx = (size_t)(bRow0 + r) * H + hCol;
        float al = 1.f / (1.f + __expf(-(acc[r] + bias)));
        float sv = al * h0c[idx] + (1.f - al) * s0c[idx];
        s0n[idx] = sv; s0nb[idx] = f2bf(sv);
      }
    }
    target += gridDim.x; grid_barrier(bar, target);

    // ---- Phase 2: h0' = tanh([x_t|s0'] W0h^T + bh0) ----
    {
      v8f acc = tile_gemm(xrow, T * D, stb + (0 + nxt) * ST, H, D, K0, W0h, aR0, wR0, lane);
      float*          h0n  = stf + (2 + nxt) * ST;
      unsigned short* h0nb = stb + (2 + nxt) * ST;
      const float bias = bh0[hCol];
      #pragma unroll
      for (int r = 0; r < 8; ++r) {
        size_t idx = (size_t)(bRow0 + r) * H + hCol;
        float hv = tanhf(acc[r] + bias);
        h0n[idx] = hv; h0nb[idx] = f2bf(hv);
      }
    }
    target += gridDim.x; grid_barrier(bar, target);

    // ---- Phase 3: alpha1 = sigmoid([h0'|s1] W1a^T + ba1); s1' = a*h1 + (1-a)*s1 ----
    {
      v8f acc = tile_gemm(stb + (2 + nxt) * ST, H, stb + (4 + cur) * ST, H, H, K1, W1a, aR0, wR0, lane);
      const float* h1c = stf + (6 + cur) * ST;
      const float* s1c = stf + (4 + cur) * ST;
      float*          s1n  = stf + (4 + nxt) * ST;
      unsigned short* s1nb = stb + (4 + nxt) * ST;
      const float bias = ba1[hCol];
      #pragma unroll
      for (int r = 0; r < 8; ++r) {
        size_t idx = (size_t)(bRow0 + r) * H + hCol;
        float al = 1.f / (1.f + __expf(-(acc[r] + bias)));
        float sv = al * h1c[idx] + (1.f - al) * s1c[idx];
        s1n[idx] = sv; s1nb[idx] = f2bf(sv);
      }
    }
    target += gridDim.x; grid_barrier(bar, target);

    // ---- Phase 4: h1' = tanh([h0'|s1'] W1h^T + bh1) ----
    {
      v8f acc = tile_gemm(stb + (2 + nxt) * ST, H, stb + (4 + nxt) * ST, H, H, K1, W1h, aR0, wR0, lane);
      float*          h1n  = stf + (6 + nxt) * ST;
      unsigned short* h1nb = stb + (6 + nxt) * ST;
      const float bias = bh1[hCol];
      #pragma unroll
      for (int r = 0; r < 8; ++r) {
        size_t idx = (size_t)(bRow0 + r) * H + hCol;
        float hv = tanhf(acc[r] + bias);
        h1n[idx] = hv; h1nb[idx] = f2bf(hv);
      }
    }
    target += gridDim.x; grid_barrier(bar, target);
  }
}

__global__ void __launch_bounds__(128)
alphat_fc(const float* __restrict__ h1f, const float* __restrict__ fcW,
          const float* __restrict__ fcb, float* __restrict__ out)
{
  int b = blockIdx.x * blockDim.x + threadIdx.x;
  if (b < B) {
    float s = 0.f;
    for (int j = 0; j < H; ++j) s += h1f[(size_t)b * H + j] * fcW[j];
    out[b] = s + fcb[0];
  }
}

__global__ void __launch_bounds__(NTHREADS)
alphat_pack(const float* __restrict__ stf, float* __restrict__ out)
{
  // T even -> final states live in parity buffer 0.
  size_t i = (size_t)blockIdx.x * blockDim.x + threadIdx.x;
  size_t stride = (size_t)gridDim.x * blockDim.x;
  for (size_t k = i; k < 4 * ST; k += stride) {
    float v;
    if      (k <     ST) v = stf[2 * ST + k];            // hidden[0] = h0
    else if (k < 2 * ST) v = stf[6 * ST + (k - ST)];     // hidden[1] = h1
    else if (k < 3 * ST) v = stf[0 * ST + (k - 2 * ST)]; // smoothed[0] = s0
    else                 v = stf[4 * ST + (k - 3 * ST)]; // smoothed[1] = s1
    out[B + k] = v;
  }
}

extern "C" void kernel_launch(void* const* d_in, const int* in_sizes, int n_in,
                              void* d_out, int out_size, void* d_ws, size_t ws_size,
                              hipStream_t stream) {
  const float* input = (const float*)d_in[0];
  const float* Wax0  = (const float*)d_in[1];
  const float* Uas0  = (const float*)d_in[2];
  const float* ba0   = (const float*)d_in[3];
  const float* Wih0  = (const float*)d_in[4];
  const float* Whh0  = (const float*)d_in[5];
  const float* bh0   = (const float*)d_in[6];
  const float* Wax1  = (const float*)d_in[7];
  const float* Uas1  = (const float*)d_in[8];
  const float* ba1   = (const float*)d_in[9];
  const float* Wih1  = (const float*)d_in[10];
  const float* Whh1  = (const float*)d_in[11];
  const float* bh1   = (const float*)d_in[12];
  const float* fcW   = (const float*)d_in[13];
  const float* fcb   = (const float*)d_in[14];

  char* ws = (char*)d_ws;
  unsigned int*   bar = (unsigned int*)(ws + WS_BAR);
  unsigned short* W0a = (unsigned short*)(ws + WS_W0A);
  unsigned short* W0h = (unsigned short*)(ws + WS_W0H);
  unsigned short* W1a = (unsigned short*)(ws + WS_W1A);
  unsigned short* W1h = (unsigned short*)(ws + WS_W1H);
  unsigned short* xbf = (unsigned short*)(ws + WS_XBF);
  float*          stf = (float*)(ws + WS_STF);
  unsigned short* stb = (unsigned short*)(ws + WS_STB);
  float* out = (float*)d_out;

  alphat_init<<<256, NTHREADS, 0, stream>>>(stf, stb, bar);
  alphat_convert_w<<<512, NTHREADS, 0, stream>>>(Wax0, Uas0, Wih0, Whh0,
                                                 Wax1, Uas1, Wih1, Whh1,
                                                 W0a, W0h, W1a, W1h);
  alphat_convert_x<<<2048, NTHREADS, 0, stream>>>(input, xbf, (size_t)B * T * D);
  alphat_rnn_persistent<<<NBLOCKS, NTHREADS, 0, stream>>>(xbf, W0a, W0h, W1a, W1h,
                                                          ba0, bh0, ba1, bh1,
                                                          stf, stb, bar);
  alphat_fc<<<1, 128, 0, stream>>>(stf + 6 * ST, fcW, fcb, out);
  alphat_pack<<<1024, NTHREADS, 0, stream>>>(stf, out);
}
